// SinkhornDistance_63660005262164
// MI455X (gfx1250) — matrix operand order repaired and checked by reading, not compiled, not run on hardware
//
#include <hip/hip_runtime.h>
#include <hip/hip_bf16.h>

typedef __attribute__((ext_vector_type(16))) _Float16 v16h;
typedef __attribute__((ext_vector_type(8)))  _Float16 v8h;
typedef __attribute__((ext_vector_type(8)))  float    v8f;

#define N_CLASS     100
#define NPAD        112   // 7 * 16
#define KPAD        128   // 4 * 32
#define NT_TILES    7
#define KC_TILES    4
#define BATCH_TOTAL 4096
#define ROWS        16
#define WAVES       4
#define THREADS     128   // 4 waves x 32
#define MAX_ITER    100
#define STABF       1e-6f
#define INV_CMAX    (1.0f / 9801.0f)   // cmax = 99^2
#define NEG_INV_EPS (-10.0f)           // -1/EPS, EPS = 0.1
#define LOG2E_F     1.4426950408889634f
#define LN2_F       0.6931471805599453f

// Raw hardware transcendentals: operands here are never denormal
// (S >= e^-10 >> 2^-126), so skip libm's denorm/inf guard sequences.
static __device__ __forceinline__ float fast_log(float x) {   // ln(x)
  return LN2_F * __builtin_amdgcn_logf(x);                    // v_log_f32
}
static __device__ __forceinline__ float fast_exp(float x) {   // e^x
  return __builtin_amdgcn_exp2f(x * LOG2E_F);                 // v_exp_f32
}

// ---------------- WMMA fragment loaders (CDNA5 wave32 layouts) ----------------

// A-matrix 16x32 f16: lane L holds M = L&15; halves e=0..7 -> K = kbase+e,
// e=8..15 -> K = kbase+16+(e-8), kbase = (L<16)?0:8.
static __device__ __forceinline__ v16h load_a_frag(const _Float16* row, int lane, int kc) {
  const _Float16* p = row + kc * 32 + ((lane & 16) ? 8 : 0);
  v8h lo = *(const v8h*)(p);
  v8h hi = *(const v8h*)(p + 16);
  return __builtin_shufflevector(lo, hi, 0,1,2,3,4,5,6,7,8,9,10,11,12,13,14,15);
}

// B-matrix 32x16 f16: pre-packed in LDS so each lane's 16 halves are contiguous.
static __device__ __forceinline__ v16h load_b_frag(const _Float16* kfrag, int lane, int nt, int kc) {
  return *(const v16h*)(kfrag + ((((nt * KC_TILES) + kc) * 32 + lane) << 4));
}

// One Sinkhorn half-step for this wave's N-tiles, state in registers.
// dst[t] <- log_marginal - (log(S) + rmax_in[m]),  S = W x K  via WMMA chain.
// Then cooperative row-max across waves and rebuild of the f16 W buffer.
static __device__ __forceinline__ void half_step(
    v8f dst[2], int nt0, int ntc,
    const float* rmax_in, float* pmax_sc, float* rmax_out,
    _Float16 (*wbuf)[KPAD], const _Float16* kfrag,
    const float (*logmu)[NPAD], const int* tgt,
    float lnu_hit, float lnu_miss, bool use_nu,
    int lane, int wave, int tid)
{
  const int n  = lane & 15;
  const int mb = (lane & 16) ? 8 : 0;
  const _Float16* arow = &wbuf[lane & 15][0];
  v16h a0 = load_a_frag(arow, lane, 0);
  v16h a1 = load_a_frag(arow, lane, 1);
  v16h a2 = load_a_frag(arow, lane, 2);
  v16h a3 = load_a_frag(arow, lane, 3);
  float pmax[8];
#pragma unroll
  for (int r = 0; r < 8; ++r) pmax[r] = -3.0e38f;

#pragma unroll
  for (int t = 0; t < 2; ++t) {
    if (t < ntc) {
      const int nt = nt0 + t;
      v16h b0 = load_b_frag(kfrag, lane, nt, 0);
      v16h b1 = load_b_frag(kfrag, lane, nt, 1);
      v16h b2 = load_b_frag(kfrag, lane, nt, 2);
      v16h b3 = load_b_frag(kfrag, lane, nt, 3);
      v8f acc = {};
      acc = __builtin_amdgcn_wmma_f32_16x16x32_f16(false, a0, false, b0, (short)0, acc, false, false);
      acc = __builtin_amdgcn_wmma_f32_16x16x32_f16(false, a1, false, b1, (short)0, acc, false, false);
      acc = __builtin_amdgcn_wmma_f32_16x16x32_f16(false, a2, false, b2, (short)0, acc, false, false);
      acc = __builtin_amdgcn_wmma_f32_16x16x32_f16(false, a3, false, b3, (short)0, acc, false, false);
      const int i = nt * 16 + n;
      v8f out;
#pragma unroll
      for (int r = 0; r < 8; ++r) {
        const int m = mb + r;
        float lm = use_nu ? ((i == tgt[m]) ? lnu_hit : lnu_miss) : logmu[m][i];
        float val = lm - (fast_log(acc[r]) + rmax_in[m]);
        out[r] = val;
        if (i < N_CLASS) pmax[r] = fmaxf(pmax[r], val);
      }
      dst[t] = out;
    }
  }

  // cross-lane max over the 16 N-columns (stays inside each 16-lane half,
  // which is exactly the C-fragment M split: lanes 0-15 -> m=r, 16-31 -> m=8+r)
#pragma unroll
  for (int r = 0; r < 8; ++r) {
    float pm = pmax[r];
    pm = fmaxf(pm, __shfl_xor(pm, 1, 32));
    pm = fmaxf(pm, __shfl_xor(pm, 2, 32));
    pm = fmaxf(pm, __shfl_xor(pm, 4, 32));
    pm = fmaxf(pm, __shfl_xor(pm, 8, 32));
    pmax[r] = pm;
  }
  if (n == 0) {     // lanes 0 and 16 publish this wave's partial row maxes
#pragma unroll
    for (int r = 0; r < 8; ++r) pmax_sc[wave * ROWS + mb + r] = pmax[r];
  }
  __syncthreads();
  if (tid < ROWS) {
    float mx = pmax_sc[tid];
#pragma unroll
    for (int w = 1; w < WAVES; ++w) mx = fmaxf(mx, pmax_sc[w * ROWS + tid]);
    rmax_out[tid] = mx;
  }
  __syncthreads();
  // rebuild W = exp(dst - rowmax) in f16, each wave writes its own columns
#pragma unroll
  for (int t = 0; t < 2; ++t) {
    if (t < ntc) {
      const int i = (nt0 + t) * 16 + n;
      if (i < N_CLASS) {
#pragma unroll
        for (int r = 0; r < 8; ++r) {
          const int m = mb + r;
          wbuf[m][i] = (_Float16)fast_exp(dst[t][r] - rmax_out[m]);
        }
      }
    }
  }
  __syncthreads();
}

// ---------------- Main kernel: 4 waves per WG, 16 batch rows, N-tiles split ----------------

__global__ __launch_bounds__(THREADS) void sinkhorn_main(
    const float* __restrict__ pred, const int* __restrict__ target,
    float* __restrict__ cost_ws, float* __restrict__ alt_ws)
{
  __shared__ __attribute__((aligned(32))) _Float16 kfrag[NT_TILES * KC_TILES * 32 * 16];
  __shared__ __attribute__((aligned(16))) _Float16 wbuf[ROWS][KPAD];
  __shared__ float  logmu[ROWS][NPAD];
  __shared__ float  u_lds[ROWS][NPAD];
  __shared__ float  v_lds[ROWS][NPAD];
  __shared__ float  rmaxU[ROWS];
  __shared__ float  rmaxV[ROWS];
  __shared__ float  pmax_sc[WAVES * ROWS];
  __shared__ float  rs_lds[WAVES * N_CLASS];
  __shared__ int    tgt[ROWS];

  const int tid  = threadIdx.x;
  const int lane = tid & 31;
  const int wave = tid >> 5;
  const int bb   = blockIdx.x * ROWS;
  const int nt0  = wave * 2;                 // N-tile split: {0,1} {2,3} {4,5} {6}
  const int ntc  = (wave < 3) ? 2 : 1;
  const float invZ = 1.0f / (1.0f + (float)N_CLASS * STABF);
  const float lnu_hit  = fast_log((1.0f + STABF) * invZ + STABF);
  const float lnu_miss = fast_log(STABF * invZ + STABF);

  // Pack K = exp(-C/EPS) into B-fragment layout once (zero padding rows/cols).
  for (int idx = tid; idx < NT_TILES * KC_TILES * 32 * 16; idx += THREADS) {
    const int e  = idx & 15;
    const int L  = (idx >> 4) & 31;
    const int kc = (idx >> 9) & 3;
    const int nt = idx >> 11;
    const int j  = kc * 32 + ((L & 16) ? 8 : 0) + ((e < 8) ? e : (e + 8));
    const int i  = nt * 16 + (L & 15);
    float val = 0.0f;
    if (i < N_CLASS && j < N_CLASS) {
      const float d = (float)(i - j);
      val = fast_exp(NEG_INV_EPS * d * d * INV_CMAX);
    }
    kfrag[idx] = (_Float16)val;
  }

  // Per-row setup: log_mu, target, alt-cost (argmax-to-argmax transport).
  if (tid < ROWS) {
    const float* prow = pred + (size_t)(bb + tid) * N_CLASS;
    float mx = -3.0e38f; int amax = 0;
    for (int i = 0; i < N_CLASS; ++i) {
      float p = prow[i];
      if (p > mx) { mx = p; amax = i; }
    }
    float se = 0.0f;
    for (int i = 0; i < N_CLASS; ++i) se += fast_exp(prow[i] - mx);
    const float inv_se = 1.0f / se;
    for (int i = 0; i < N_CLASS; ++i) {
      float mui = (fast_exp(prow[i] - mx) * inv_se + STABF) * invZ;
      logmu[tid][i] = fast_log(mui + STABF);
    }
    for (int i = N_CLASS; i < NPAD; ++i) logmu[tid][i] = 0.0f;
    const int tg = target[bb + tid];
    tgt[tid] = tg;
    const float dd = (float)(amax - tg);
    alt_ws[bb + tid] = dd * dd * INV_CMAX;
    rmaxV[tid] = 0.0f;                       // v0 = 0 -> rowmax(v) = 0
  }
  // W for first half-step: exp(0 - 0) = 1 on real classes, 0 on padding.
  for (int idx = tid; idx < ROWS * KPAD; idx += THREADS) {
    const int k = idx & (KPAD - 1);
    wbuf[idx >> 7][k] = (k < N_CLASS) ? (_Float16)1.0f : (_Float16)0.0f;
  }
  __syncthreads();

  // 100 Sinkhorn iterations; u,v live in registers (WMMA C-fragment layout).
  v8f uregs[2], vregs[2];
  for (int it = 0; it < MAX_ITER; ++it) {
    half_step(uregs, nt0, ntc, rmaxV, pmax_sc, rmaxU, wbuf, kfrag,
              logmu, tgt, lnu_hit, lnu_miss, false, lane, wave, tid);
    half_step(vregs, nt0, ntc, rmaxU, pmax_sc, rmaxV, wbuf, kfrag,
              logmu, tgt, lnu_hit, lnu_miss, true,  lane, wave, tid);
  }

  // Dump final u, v to LDS for the transport-plan cost phase.
  {
    const int n  = lane & 15;
    const int mb = (lane & 16) ? 8 : 0;
#pragma unroll
    for (int t = 0; t < 2; ++t) {
      if (t < ntc) {
        const int i = (nt0 + t) * 16 + n;
        if (i < N_CLASS) {
#pragma unroll
          for (int r = 0; r < 8; ++r) {
            u_lds[mb + r][i] = uregs[t][r];
            v_lds[mb + r][i] = vregs[t][r];
          }
        }
      }
    }
  }
  __syncthreads();

  // Transport-plan cost with row/col renormalization; 4 rows per wave.
  float* rs = &rs_lds[wave * N_CLASS];
  for (int mm = 0; mm < 4; ++mm) {
    const int m = wave * 4 + mm;
    for (int i = lane; i < N_CLASS; i += 32) {
      const float ui = u_lds[m][i];
      float s = 0.0f;
      for (int j = 0; j < N_CLASS; ++j) {
        const float d = (float)(i - j);
        s += fast_exp(ui + v_lds[m][j] + NEG_INV_EPS * d * d * INV_CMAX) + STABF;
      }
      rs[i] = s;
    }
    __syncthreads();
    for (int i = lane; i < N_CLASS; i += 32)
      rs[i] = (fast_exp(logmu[m][i]) - STABF) / rs[i];
    __syncthreads();
    float part = 0.0f;
    for (int j = lane; j < N_CLASS; j += 32) {
      const float vj = v_lds[m][j];
      float cs = 0.0f, ts = 0.0f;
      for (int i = 0; i < N_CLASS; ++i) {
        const float d   = (float)(i - j);
        const float Cij = d * d * INV_CMAX;
        const float p0  = fast_exp(u_lds[m][i] + vj + NEG_INV_EPS * Cij) + STABF;
        const float w   = p0 * rs[i];
        cs += w; ts += w * Cij;
      }
      const float nuj = ((j == tgt[m]) ? (1.0f + STABF) : STABF) * invZ;
      part += nuj * ts / cs;
    }
    for (int off = 16; off > 0; off >>= 1) part += __shfl_xor(part, off, 32);
    if (lane == 0) cost_ws[bb + m] = part;
    __syncthreads();
  }
}

// ---------------- NaN-any check + mean ----------------

__global__ __launch_bounds__(256) void sinkhorn_reduce(
    const float* __restrict__ cost, const float* __restrict__ alt,
    float* __restrict__ out)
{
  __shared__ float ssum[256];
  __shared__ int   snan[256];
  const int t = threadIdx.x;
  int anyn = 0;
  for (int i = t; i < BATCH_TOTAL; i += 256) anyn |= __builtin_isnan(cost[i]) ? 1 : 0;
  snan[t] = anyn;
  __syncthreads();
  for (int o = 128; o > 0; o >>= 1) { if (t < o) snan[t] |= snan[t + o]; __syncthreads(); }
  const int anynan = snan[0];
  __syncthreads();
  float s = 0.0f;
  for (int i = t; i < BATCH_TOTAL; i += 256) s += anynan ? alt[i] : cost[i];
  ssum[t] = s;
  __syncthreads();
  for (int o = 128; o > 0; o >>= 1) { if (t < o) ssum[t] += ssum[t + o]; __syncthreads(); }
  if (t == 0) out[0] = ssum[0] * (1.0f / (float)BATCH_TOTAL);
}

extern "C" void kernel_launch(void* const* d_in, const int* in_sizes, int n_in,
                              void* d_out, int out_size, void* d_ws, size_t ws_size,
                              hipStream_t stream) {
  (void)in_sizes; (void)n_in; (void)out_size; (void)ws_size;
  const float* pred   = (const float*)d_in[0];
  const int*   target = (const int*)d_in[1];
  float* cost_ws = (float*)d_ws;
  float* alt_ws  = cost_ws + BATCH_TOTAL;
  sinkhorn_main<<<BATCH_TOTAL / ROWS, THREADS, 0, stream>>>(pred, target, cost_ws, alt_ws);
  sinkhorn_reduce<<<1, 256, 0, stream>>>(cost_ws, alt_ws, (float*)d_out);
}
